// MultiheadAttention_1202590842968
// MI455X (gfx1250) — matrix operand order repaired and checked by reading, not compiled, not run on hardware
//
#include <hip/hip_runtime.h>
#include <hip/hip_bf16.h>

// MultiheadAttention forward for MI455X (gfx1250), bf16 WMMA with fp32 accum.
// T=2048, B=2, E=1024, H=16, D=64. Outputs: out [T,B,E] fp32 then
// attn_weights [B*H,T,T] fp32, concatenated in d_out.

typedef __bf16 bf16_t;
typedef __attribute__((ext_vector_type(16))) __bf16 v16bf;
typedef __attribute__((ext_vector_type(4)))  __bf16 v4bf;
typedef __attribute__((ext_vector_type(8)))  float  v8f;
typedef __attribute__((ext_vector_type(4)))  float  v4f;
typedef __attribute__((ext_vector_type(4)))  unsigned int v4u;
typedef __attribute__((ext_vector_type(8)))  int    v8i;
typedef __attribute__((ext_vector_type(4)))  int    v4i;

#define T_SEQ  2048
#define BATCH  2
#define EMB    1024
#define HEADS  16
#define HDIM   64
#define BHDIM  (BATCH * HEADS)         // 32
#define M_ROWS (T_SEQ * BATCH)         // 4096
#define SSTR   2056                    // LDS row stride (floats), bank-skewed

// ---- WMMA helpers -----------------------------------------------------------

static __device__ __forceinline__ v8f wmma_bf16(v16bf a, v16bf b, v8f c) {
  // 8 args: (neg_a, A, neg_b, B, c_mod, C, reuse_a, reuse_b)
  return __builtin_amdgcn_wmma_f32_16x16x32_bf16(false, a, false, b,
                                                 (short)0, c, false, false);
}

// A-fragment (16x32) from an fp32 row (LDS), converting to bf16.
// ISA layout: lane L holds row M=L&15; lanes 0-15 get K {0..7,16..23},
// lanes 16-31 get K {8..15,24..31} (relative to k0).
static __device__ __forceinline__ v16bf a_frag_from_f32(const float* row, int k0, int lane) {
  const int hs = (lane >> 4) & 1;
  const float* p0 = row + k0 + hs * 8;
  v16bf a;
#pragma unroll
  for (int i = 0; i < 8; ++i) a[i] = (__bf16)p0[i];
#pragma unroll
  for (int i = 0; i < 8; ++i) a[8 + i] = (__bf16)p0[16 + i];
  return a;
}

// A-fragment from a bf16 row (global or LDS).
static __device__ __forceinline__ v16bf a_frag_from_bf16(const bf16_t* row, int k0, int lane) {
  const int hs = (lane >> 4) & 1;
  const bf16_t* p0 = row + k0 + hs * 8;
  v16bf a;
#pragma unroll
  for (int i = 0; i < 8; ++i) a[i] = p0[i];
#pragma unroll
  for (int i = 0; i < 8; ++i) a[8 + i] = p0[16 + i];
  return a;
}

// B-fragment (32x16 bf16): lane L holds column N=n0+(L&15); lanes 0-15 get
// K {0..7,16..23}, lanes 16-31 get K {8..15,24..31} (relative to k0).
static __device__ __forceinline__ v16bf b_frag(const bf16_t* Bmat, size_t ldb,
                                               int k0, int n0, int lane) {
  const int hs = (lane >> 4) & 1;
  const bf16_t* p = Bmat + (size_t)(k0 + hs * 8) * ldb + n0 + (lane & 15);
  v16bf b;
#pragma unroll
  for (int i = 0; i < 8; ++i) b[i] = p[(size_t)i * ldb];
#pragma unroll
  for (int i = 0; i < 8; ++i) b[8 + i] = p[(size_t)(16 + i) * ldb];
  return b;
}

// ---- Kernel 0: query fp32 -> bf16 ------------------------------------------
__global__ void __launch_bounds__(256)
xconv_kernel(const float* __restrict__ X, bf16_t* __restrict__ Xbf) {
  const size_t i = ((size_t)blockIdx.x * 256 + threadIdx.x) * 4;
  const v4f v = *(const v4f*)(X + i);
  v4bf o;
  o[0] = (__bf16)v[0]; o[1] = (__bf16)v[1];
  o[2] = (__bf16)v[2]; o[3] = (__bf16)v[3];
  *(v4bf*)(Xbf + i) = o;
}

// ---- Kernel 1: weight transpose + fp32 -> bf16 ------------------------------
// WT[k][n] = W[n][k]; wt_all holds [wq,wk,wv,wo] transposed, each E*E bf16.
__global__ void __launch_bounds__(256)
wt_kernel(const float* __restrict__ wq, const float* __restrict__ wk,
          const float* __restrict__ wv, const float* __restrict__ wo,
          bf16_t* __restrict__ wt_all) {
  const float* W = (blockIdx.z == 0) ? wq : (blockIdx.z == 1) ? wk
                 : (blockIdx.z == 2) ? wv : wo;
  bf16_t* WT = wt_all + (size_t)blockIdx.z * EMB * EMB;
  __shared__ float tile[32][33];
  const int tx = threadIdx.x & 31;
  const int ty = threadIdx.x >> 5;            // 8 rows of 32
  const int n0 = blockIdx.x * 32, k0 = blockIdx.y * 32;
#pragma unroll
  for (int j = 0; j < 4; ++j)
    tile[ty + j * 8][tx] = W[(size_t)(n0 + ty + j * 8) * EMB + k0 + tx];
  __syncthreads();
#pragma unroll
  for (int j = 0; j < 4; ++j)
    WT[(size_t)(k0 + ty + j * 8) * EMB + n0 + tx] = (__bf16)tile[tx][ty + j * 8];
}

// ---- Kernel 2: QKV projection ----------------------------------------------
// The 16x32 A tile is shared by all 8 waves of the block: stage it in LDS with
// async global->LDS DMA (ASYNCcnt), double-buffered to overlap with WMMA.
__global__ void __launch_bounds__(256)
qkv_kernel(const bf16_t* __restrict__ Xbf, const bf16_t* __restrict__ wt_all,
           const float* __restrict__ bq, const float* __restrict__ bk,
           const float* __restrict__ bv,
           bf16_t* __restrict__ Qh, bf16_t* __restrict__ KhT,
           bf16_t* __restrict__ Vh) {
  __shared__ bf16_t Atile[2][16][40];          // 32 valid cols, padded stride
  const int z = blockIdx.z;
  const bf16_t* WT  = wt_all + (size_t)z * EMB * EMB;
  const float* bias = (z == 0) ? bq : (z == 1) ? bk : bv;
  const int tid  = threadIdx.x;
  const int lane = tid & 31;
  const int wave = tid >> 5;
  const int m0 = blockIdx.y * 16;
  const int n0 = (blockIdx.x * 8 + wave) * 16;

  // Async stage of A tile: 16 rows x 64B = 64 x b128 chunks (threads 0..63).
  auto kick = [&](int buf, int k0) {
    const int row = tid >> 2, part = tid & 3;
    const unsigned int laddr =
        (unsigned int)(size_t)&Atile[buf][row][part * 8];
    const bf16_t* g = Xbf + (size_t)(m0 + row) * EMB + k0 + part * 8;
    asm volatile("global_load_async_to_lds_b128 %0, %1, off"
                 :: "v"(laddr), "v"(g) : "memory");
  };

  if (tid < 64) kick(0, 0);

  v8f acc = {};
  for (int k0 = 0; k0 < EMB; k0 += 32) {
    const int buf = (k0 >> 5) & 1;
    const bool more = (k0 + 32) < EMB;
    __syncthreads();                       // prior reads of buf^1 are done
    if (tid < 64) {
      if (more) {
        kick(buf ^ 1, k0 + 32);
        asm volatile("s_wait_asynccnt 1" ::: "memory");  // current buf landed
      } else {
        asm volatile("s_wait_asynccnt 0" ::: "memory");
      }
    }
    __syncthreads();                       // Atile[buf] visible to all waves
    if (more)                              // pull next B rows into WGP$
      __builtin_prefetch(WT + (size_t)(k0 + 32 + lane) * EMB + n0, 0, 3);
    v16bf a = a_frag_from_bf16(&Atile[buf][lane & 15][0], 0, lane);
    v16bf b = b_frag(WT, EMB, k0, n0, lane);
    acc = wmma_bf16(a, b, acc);
  }

  const int n  = n0 + (lane & 15);
  const float bn = bias[n];
  const int h = n >> 6, d = n & 63;
  const float scal = (z == 0) ? 0.125f : 1.0f;     // D^-0.5 = 1/8
#pragma unroll
  for (int r = 0; r < 8; ++r) {
    const int m = m0 + r + ((lane >> 4) << 3);
    const int t = m >> 1, bb = m & 1;              // m = t*B + b, B=2
    const int nbh = bb * HEADS + h;
    const float val = (acc[r] + bn) * scal;
    if (z == 0)      Qh [((size_t)nbh * T_SEQ + t) * HDIM + d] = (__bf16)val;
    else if (z == 1) KhT[((size_t)nbh * HDIM + d) * T_SEQ + t] = (__bf16)val;
    else             Vh [((size_t)nbh * T_SEQ + t) * HDIM + d] = (__bf16)val;
  }
}

// ---- Kernel 3: attention (scores -> softmax in LDS -> P@V) ------------------
// One block per (16-row strip t0, head-batch nbh). 8 waves / 256 threads.
// The Q strip (16x64 bf16, reused by every wave for all 128 column tiles) is
// staged into LDS with one Tensor-Data-Mover descriptor (TENSORcnt).
__global__ void __launch_bounds__(256)
attn_kernel(const bf16_t* __restrict__ Qh, const bf16_t* __restrict__ KhT,
            const bf16_t* __restrict__ Vh,
            float* __restrict__ attnW,      // [BH][T][T] fp32 (d_out part 2)
            bf16_t* __restrict__ attnM) {   // [M_ROWS][EMB] bf16
  __shared__ float  S[16 * SSTR];           // 16 x 2048 score strip (~128 KB)
  __shared__ float  red[16 * 16];
  __shared__ float  rowmax[16];
  __shared__ float  rowsum[16];
  __shared__ bf16_t Qs[16][HDIM];           // TDM-staged Q strip (2 KB)

  const int nbh  = blockIdx.y;
  const int t0   = blockIdx.x * 16;
  const int lane = threadIdx.x & 31;
  const int wave = threadIdx.x >> 5;

  // --- TDM: Qs[0:16][0:64] <- Qh[nbh][t0:t0+16][0:64] (2-D tile, bf16) ------
  if (wave == 0) {
    const size_t gaddr = (size_t)(Qh + ((size_t)nbh * T_SEQ + t0) * HDIM);
    const unsigned int laddr = (unsigned int)(size_t)&Qs[0][0];
    v4u g0;
    g0[0] = 1u;                                    // count=1, user descriptor
    g0[1] = laddr;                                 // lds_addr
    g0[2] = (unsigned int)(gaddr & 0xffffffffu);   // global_addr[31:0]
    g0[3] = (unsigned int)((gaddr >> 32) & 0x01ffffffu) | (2u << 30); // type=2
    v8i g1 = {};
    g1[0] = (1 << 16);                             // data_size = 1 (2 bytes)
    g1[1] = (HDIM  & 0xffff) << 16;                // tensor_dim0[15:0]
    g1[2] = (T_SEQ & 0xffff) << 16;                // tensor_dim1[15:0]
    g1[3] = (HDIM  & 0xffff) << 16;                // tile_dim0 = 64
    g1[4] = 16;                                    // tile_dim1 = 16
    g1[5] = HDIM;                                  // tensor_dim0_stride = 64
    g1[6] = 0; g1[7] = 0;
    v4i gz4 = {};
    v8i gz8 = {};
    __builtin_amdgcn_tensor_load_to_lds(g0, g1, gz4, gz4, gz8, 0);
    __builtin_amdgcn_s_wait_tensorcnt(0);
  }
  __syncthreads();

  // Phase 1: S[0:16, 0:2048] = Q(strip) @ K^T, via WMMA (K-dim = HDIM = 64).
  const v16bf a0 = a_frag_from_bf16(&Qs[lane & 15][0], 0,  lane);
  const v16bf a1 = a_frag_from_bf16(&Qs[lane & 15][0], 32, lane);
  for (int ct = 0; ct < 16; ++ct) {
    const int s0 = (wave * 16 + ct) * 16;
    v8f acc = {};
    acc = wmma_bf16(a0, b_frag(KhT + (size_t)nbh * HDIM * T_SEQ, T_SEQ, 0,  s0, lane), acc);
    acc = wmma_bf16(a1, b_frag(KhT + (size_t)nbh * HDIM * T_SEQ, T_SEQ, 32, s0, lane), acc);
#pragma unroll
    for (int r = 0; r < 8; ++r)
      S[(r + ((lane >> 4) << 3)) * SSTR + s0 + (lane & 15)] = acc[r];
  }
  __syncthreads();

  // Phase 2: row-wise softmax fully in LDS; stream normalized weights to HBM.
  {
    const int row = threadIdx.x >> 4;       // 16 rows x 16 threads
    const int c   = threadIdx.x & 15;
    float m = -3.402823466e38f;
    for (int s = c; s < T_SEQ; s += 16) m = fmaxf(m, S[row * SSTR + s]);
    red[row * 16 + c] = m;
    __syncthreads();
    if (c == 0) {
      float mm = red[row * 16];
#pragma unroll
      for (int i = 1; i < 16; ++i) mm = fmaxf(mm, red[row * 16 + i]);
      rowmax[row] = mm;
    }
    __syncthreads();
    const float mm = rowmax[row];
    float sum = 0.f;
    for (int s = c; s < T_SEQ; s += 16) {
      const float e = __expf(S[row * SSTR + s] - mm);
      S[row * SSTR + s] = e;
      sum += e;
    }
    red[row * 16 + c] = sum;
    __syncthreads();
    if (c == 0) {
      float ss = 0.f;
#pragma unroll
      for (int i = 0; i < 16; ++i) ss += red[row * 16 + i];
      rowsum[row] = ss;
    }
    __syncthreads();
    const float inv = 1.0f / rowsum[row];
    float* orow = attnW + ((size_t)nbh * T_SEQ + t0 + row) * T_SEQ;
    for (int s = c; s < T_SEQ; s += 16) {
      const float p = S[row * SSTR + s] * inv;
      S[row * SSTR + s] = p;
      orow[s] = p;                          // 537 MB stream: the BW floor
    }
  }
  __syncthreads();

  // Phase 3: O(strip) = P @ V. Each wave covers a 256-wide s-range; 16x64 tile.
  v8f accd[4] = {};
  const float* prow = &S[(lane & 15) * SSTR];
  for (int i = 0; i < 8; ++i) {
    const int ss = wave * 256 + i * 32;
    v16bf a = a_frag_from_f32(prow, ss, lane);   // fp32 LDS -> bf16 frag
#pragma unroll
    for (int dt = 0; dt < 4; ++dt) {
      v16bf b = b_frag(Vh + (size_t)nbh * T_SEQ * HDIM, HDIM, ss, dt * 16, lane);
      accd[dt] = wmma_bf16(a, b, accd[dt]);
    }
  }
  __syncthreads();                          // everyone done reading S
  // Stash per-wave 16x64 partials in (now-free) S, then sum across waves.
#pragma unroll
  for (int dt = 0; dt < 4; ++dt)
#pragma unroll
    for (int r = 0; r < 8; ++r)
      S[wave * 1024 + (r + ((lane >> 4) << 3)) * 64 + dt * 16 + (lane & 15)] =
          accd[dt][r];
  __syncthreads();
  {
    const int e = threadIdx.x * 4;          // 256 threads x 4 = 1024 elems
    const int r = e >> 6, d0 = e & 63;
    float v0 = 0.f, v1 = 0.f, v2 = 0.f, v3 = 0.f;
#pragma unroll
    for (int w = 0; w < 8; ++w) {
      const float* p = &S[w * 1024 + r * 64 + d0];
      v0 += p[0]; v1 += p[1]; v2 += p[2]; v3 += p[3];
    }
    const int bb = nbh >> 4, h = nbh & 15;  // nbh = b*H + h
    bf16_t* o = attnM + ((size_t)(t0 + r) * BATCH + bb) * EMB + h * HDIM + d0;
    o[0] = (__bf16)v0; o[1] = (__bf16)v1; o[2] = (__bf16)v2; o[3] = (__bf16)v3;
  }
}

// ---- Kernel 4: output projection -------------------------------------------
__global__ void __launch_bounds__(256)
outproj_kernel(const bf16_t* __restrict__ A, const bf16_t* __restrict__ WoT,
               const float* __restrict__ bo, float* __restrict__ out) {
  const int lane = threadIdx.x & 31;
  const int wave = threadIdx.x >> 5;
  const int m0 = blockIdx.y * 16;
  const int n0 = (blockIdx.x * 8 + wave) * 16;
  const bf16_t* arow = A + (size_t)(m0 + (lane & 15)) * EMB;

  v8f acc = {};
  for (int k0 = 0; k0 < EMB; k0 += 32) {
    if (k0 + 32 < EMB)
      __builtin_prefetch(WoT + (size_t)(k0 + 32 + lane) * EMB + n0, 0, 3);
    v16bf a = a_frag_from_bf16(arow, k0, lane);
    v16bf b = b_frag(WoT, EMB, k0, n0, lane);
    acc = wmma_bf16(a, b, acc);
  }
  const int n = n0 + (lane & 15);
  const float bn = bo[n];
#pragma unroll
  for (int r = 0; r < 8; ++r) {
    const int m = m0 + r + ((lane >> 4) << 3);
    out[(size_t)m * EMB + n] = acc[r] + bn;
  }
}

// ---- Launch -----------------------------------------------------------------
extern "C" void kernel_launch(void* const* d_in, const int* in_sizes, int n_in,
                              void* d_out, int out_size, void* d_ws, size_t ws_size,
                              hipStream_t stream) {
  const float* query = (const float*)d_in[0];
  const float* wq = (const float*)d_in[1];
  const float* bq = (const float*)d_in[2];
  const float* wk = (const float*)d_in[3];
  const float* bk = (const float*)d_in[4];
  const float* wv = (const float*)d_in[5];
  const float* bv = (const float*)d_in[6];
  const float* wo = (const float*)d_in[7];
  const float* bo = (const float*)d_in[8];

  float* out   = (float*)d_out;                        // [T,B,E]
  float* attnW = out + (size_t)M_ROWS * EMB;           // [BH,T,T]

  // Workspace map (bf16, 48 MB total):
  //   [0,8M):   4 transposed weight matrices (q,k,v,o)
  //   [8M,40M): Qh, KhT, Vh, attnM (8 MB each)
  //   [40M,48M): Xbf (query in bf16)
  char* ws = (char*)d_ws;
  bf16_t* wt_all = (bf16_t*)ws;
  bf16_t* Qh     = (bf16_t*)(ws + (size_t) 8 * 1024 * 1024);
  bf16_t* KhT    = (bf16_t*)(ws + (size_t)16 * 1024 * 1024);
  bf16_t* Vh     = (bf16_t*)(ws + (size_t)24 * 1024 * 1024);
  bf16_t* attnM  = (bf16_t*)(ws + (size_t)32 * 1024 * 1024);
  bf16_t* Xbf    = (bf16_t*)(ws + (size_t)40 * 1024 * 1024);
  bf16_t* WoT    = wt_all + (size_t)3 * EMB * EMB;

  xconv_kernel  <<<dim3(4096, 1, 1), 256, 0, stream>>>(query, Xbf);
  wt_kernel     <<<dim3(32, 32, 4),  256, 0, stream>>>(wq, wk, wv, wo, wt_all);
  qkv_kernel    <<<dim3(8, 256, 3),  256, 0, stream>>>(Xbf, wt_all, bq, bk, bv,
                                                       Qh, KhT, Vh);
  attn_kernel   <<<dim3(128, 32, 1), 256, 0, stream>>>(Qh, KhT, Vh, attnW, attnM);
  outproj_kernel<<<dim3(8, 256, 1),  256, 0, stream>>>(attnM, WoT, bo, out);
}
